// TripletLossInLatentSpaceCosSimv2_62113817035257
// MI455X (gfx1250) — matrix operand order repaired and checked by reading, not compiled, not run on hardware
//
#include <hip/hip_runtime.h>

// CDNA5 / gfx1250, wave32. WMMA f16->f32 GEMM, async-to-LDS triple-buffer ring.

typedef __attribute__((ext_vector_type(16))) _Float16 v16h;
typedef __attribute__((ext_vector_type(8)))  _Float16 v8h;
typedef __attribute__((ext_vector_type(8)))  float    v8f;

#define D_DIM     256        // feature dim (reference)
#define BS_LDH    264        // padded LDS row stride in halves (16B pad -> conflict-free b128)
#define TILE_H    (16 * BS_LDH)          // halves per LDS tile buffer
#define BUF_BYTES (TILE_H * 2)           // 8448 bytes per buffer
#define J_SLICES  8          // split of the j (neg-row) dimension across blockIdx.y

// ---- CDNA5 async global->LDS DMA (ASYNCcnt-tracked), per ISA 15.18.3 ----
__device__ __forceinline__ void async_b128(unsigned lds_addr, const void* gaddr) {
    // vdst = LDS byte address (low 32 bits of generic shared pointer), vaddr = 64-bit global
    asm volatile("global_load_async_to_lds_b128 %0, %1, off"
                 :: "v"(lds_addr), "v"(gaddr)
                 : "memory");
}
__device__ __forceinline__ void wait_async_le2() {
    asm volatile("s_wait_asynccnt 0x2" ::: "memory");
}
__device__ __forceinline__ void wait_async_le0() {
    asm volatile("s_wait_asynccnt 0x0" ::: "memory");
}

// ---------------------------------------------------------------------------
// Kernel 1: row-wise norms, D1 = exp(cos(anchor,pos)), write f16 a_hat/n_hat
// ---------------------------------------------------------------------------
__global__ __launch_bounds__(256) void tl_normalize_kernel(
    const float* __restrict__ anchor,
    const float* __restrict__ pos,
    const float* __restrict__ neg,
    _Float16* __restrict__ a_hat,
    _Float16* __restrict__ n_hat,
    float* __restrict__ D1,
    int Bn)
{
    const int wave = threadIdx.x >> 5;
    const int lane = threadIdx.x & 31;
    const int row  = blockIdx.x * (blockDim.x >> 5) + wave;
    if (row >= Bn) return;

    const int base = row * D_DIM + lane * 8;
    float4 a0 = *(const float4*)(anchor + base);
    float4 a1 = *(const float4*)(anchor + base + 4);
    float4 p0 = *(const float4*)(pos    + base);
    float4 p1 = *(const float4*)(pos    + base + 4);
    float4 g0 = *(const float4*)(neg    + base);
    float4 g1 = *(const float4*)(neg    + base + 4);

    float aa = a0.x*a0.x + a0.y*a0.y + a0.z*a0.z + a0.w*a0.w
             + a1.x*a1.x + a1.y*a1.y + a1.z*a1.z + a1.w*a1.w;
    float pp = p0.x*p0.x + p0.y*p0.y + p0.z*p0.z + p0.w*p0.w
             + p1.x*p1.x + p1.y*p1.y + p1.z*p1.z + p1.w*p1.w;
    float ap = a0.x*p0.x + a0.y*p0.y + a0.z*p0.z + a0.w*p0.w
             + a1.x*p1.x + a1.y*p1.y + a1.z*p1.z + a1.w*p1.w;
    float gg = g0.x*g0.x + g0.y*g0.y + g0.z*g0.z + g0.w*g0.w
             + g1.x*g1.x + g1.y*g1.y + g1.z*g1.z + g1.w*g1.w;

    #pragma unroll
    for (int m = 1; m < 32; m <<= 1) {
        aa += __shfl_xor(aa, m, 32);
        pp += __shfl_xor(pp, m, 32);
        ap += __shfl_xor(ap, m, 32);
        gg += __shfl_xor(gg, m, 32);
    }

    const float an = sqrtf(aa);
    const float pn = sqrtf(pp);
    const float gn = sqrtf(gg);

    if (lane == 0) {
        const float cosv = ap / fmaxf(an * pn, 1e-8f);
        D1[row] = __expf(cosv);            // tau = 1
    }

    const float ia = 1.0f / fmaxf(an, 1e-12f);
    const float ig = 1.0f / fmaxf(gn, 1e-12f);

    v8h ah, nh;
    ah[0] = (_Float16)(a0.x * ia); ah[1] = (_Float16)(a0.y * ia);
    ah[2] = (_Float16)(a0.z * ia); ah[3] = (_Float16)(a0.w * ia);
    ah[4] = (_Float16)(a1.x * ia); ah[5] = (_Float16)(a1.y * ia);
    ah[6] = (_Float16)(a1.z * ia); ah[7] = (_Float16)(a1.w * ia);
    nh[0] = (_Float16)(g0.x * ig); nh[1] = (_Float16)(g0.y * ig);
    nh[2] = (_Float16)(g0.z * ig); nh[3] = (_Float16)(g0.w * ig);
    nh[4] = (_Float16)(g1.x * ig); nh[5] = (_Float16)(g1.y * ig);
    nh[6] = (_Float16)(g1.z * ig); nh[7] = (_Float16)(g1.w * ig);

    *(v8h*)(a_hat + base) = ah;
    *(v8h*)(n_hat + base) = nh;
}

// ---------------------------------------------------------------------------
// Kernel 2: D2part[slice][i] = sum_{j in slice} exp(<a_hat_i, n_hat_j>)
//   grid = (Bn/128, J_SLICES); 8 waves/block; A tile pinned in 64 VGPRs/wave.
//   B tiles (16x256 f16) stream through a 3-deep LDS ring filled by
//   global_load_async_to_lds_b128; ONE barrier per tile:
//     prefetch(t+1) into buf (t+1)%3 is issued only after barrier(t-1), which
//     all waves pass only after finishing compute(t-2) = last reader of that
//     buffer. Async loads complete in-order per wave, so asynccnt<=2 retires
//     exactly the current tile's pair with the next tile's pair in flight.
// Frag layouts per ISA 7.12.2:
//   A (16x32): lane=M%16; elems 0..7 -> K=kh..kh+7, 8..15 -> K=16+kh.., kh=8*(lane/16)
//   B (32x16): lane=N%16; elems 0..7 -> K=kh..kh+7, 8..15 -> K=16+kh.. (same as A)
//   C/D:       elem r -> M = r + 8*(lane/16), N = lane%16
// ---------------------------------------------------------------------------
__global__ __launch_bounds__(256) void tl_gemm_exp_kernel(
    const _Float16* __restrict__ a_hat,
    const _Float16* __restrict__ n_hat,
    float* __restrict__ D2part,
    int Bn)
{
    __shared__ _Float16 Bs[3][TILE_H];

    const int tid     = threadIdx.x;
    const int wave    = tid >> 5;
    const int lane    = tid & 31;
    const int rowBase = blockIdx.x * 128 + wave * 16;
    const int m       = lane & 15;
    const int hiGrp   = lane >> 4;      // 0 or 1
    const int kh      = hiGrp * 8;      // A sub-K offset
    const int kb      = hiGrp * 16;     // unused; B reads at kh like A

    // staging addresses for this thread (two 16B chunks of the 8KB tile)
    const int c0 = tid,        j0 = c0 >> 5, k0 = (c0 & 31) * 8;
    const int c1 = tid + 256,  j1 = c1 >> 5, k1 = (c1 & 31) * 8;
    const unsigned stage0 = (unsigned)(uintptr_t)&Bs[0][j0 * BS_LDH + k0];
    const unsigned stage1 = (unsigned)(uintptr_t)&Bs[0][j1 * BS_LDH + k1];

    // --- pin the 16x256 A tile in registers (8 k-tiles of v16h) ---
    v16h Areg[8];
    #pragma unroll
    for (int kt = 0; kt < 8; ++kt) {
        const _Float16* ap = a_hat + (size_t)(rowBase + m) * D_DIM + kt * 32;
        v8h lo = *(const v8h*)(ap + kh);
        v8h hi = *(const v8h*)(ap + 16 + kh);
        Areg[kt] = __builtin_shufflevector(lo, hi,
                     0,1,2,3,4,5,6,7,8,9,10,11,12,13,14,15);
    }

    v8f sumv = {0.f,0.f,0.f,0.f,0.f,0.f,0.f,0.f};

    const int nT  = Bn >> 4;
    const int tps = nT / J_SLICES;             // tiles per slice
    const int jt0 = blockIdx.y * tps;

    // prefetch tile 0 of this slice into buffer 0
    {
        const _Float16* src = n_hat + (size_t)(jt0 << 4) * D_DIM;
        async_b128(stage0, src + j0 * D_DIM + k0);
        async_b128(stage1, src + j1 * D_DIM + k1);
    }

    int bufRd = 0;   // byte offset factor of buffer holding tile t
    int bufWr = 1;   // buffer receiving tile t+1
    for (int t = 0; t < tps; ++t) {
        if (t + 1 < tps) {
            const _Float16* src = n_hat + (size_t)((jt0 + t + 1) << 4) * D_DIM;
            const unsigned woff = (unsigned)(bufWr * BUF_BYTES);
            async_b128(stage0 + woff, src + j0 * D_DIM + k0);
            async_b128(stage1 + woff, src + j1 * D_DIM + k1);
            wait_async_le2();
        } else {
            wait_async_le0();
        }
        __syncthreads();   // current tile in LDS + all waves done reading buf(t-2)

        const _Float16* bbase = &Bs[0][m * BS_LDH + kh] + bufRd * TILE_H;

        // software-pipelined B fragments: load kt+1 while WMMA kt executes
        v8h blo = *(const v8h*)(bbase);
        v8h bhi = *(const v8h*)(bbase + 16);
        v16h Bcur = __builtin_shufflevector(blo, bhi,
                      0,1,2,3,4,5,6,7,8,9,10,11,12,13,14,15);
        v8f c = {0.f,0.f,0.f,0.f,0.f,0.f,0.f,0.f};
        #pragma unroll
        for (int kt = 0; kt < 8; ++kt) {
            v16h Bnxt = Bcur;
            if (kt < 7) {
                const _Float16* bp = bbase + (kt + 1) * 32;
                v8h nlo = *(const v8h*)(bp);
                v8h nhi = *(const v8h*)(bp + 16);
                Bnxt = __builtin_shufflevector(nlo, nhi,
                         0,1,2,3,4,5,6,7,8,9,10,11,12,13,14,15);
            }
            c = __builtin_amdgcn_wmma_f32_16x16x32_f16(
                    /*neg_a=*/false, Areg[kt],
                    /*neg_b=*/false, Bcur,
                    /*c_mod=*/(short)0, c,
                    /*reuse_a=*/false, /*reuse_b=*/false);
            Bcur = Bnxt;
        }

        // epilogue: exp of each score, accumulate per-row partial sums
        #pragma unroll
        for (int r = 0; r < 8; ++r) sumv[r] += __expf(c[r]);

        bufRd = (bufRd == 2) ? 0 : bufRd + 1;
        bufWr = (bufWr == 2) ? 0 : bufWr + 1;
    }

    // --- reduce over N (16 lanes per group), write partial D2 rows ---
    float* dst = D2part + (size_t)blockIdx.y * Bn;
    #pragma unroll
    for (int r = 0; r < 8; ++r) {
        float v = sumv[r];
        v += __shfl_xor(v, 1, 32);
        v += __shfl_xor(v, 2, 32);
        v += __shfl_xor(v, 4, 32);
        v += __shfl_xor(v, 8, 32);
        if (lane == 0)  dst[rowBase + r]     = v;   // rows M=0..7   (group 0)
        if (lane == 16) dst[rowBase + 8 + r] = v;   // rows M=8..15  (group 1)
    }
}

// ---------------------------------------------------------------------------
// Kernel 3: loss = mean_i log(1 + D2[i]/D1[i]),  D2 = fixed-order slice sum
// ---------------------------------------------------------------------------
__global__ __launch_bounds__(256) void tl_finalize_kernel(
    const float* __restrict__ D1,
    const float* __restrict__ D2part,
    float* __restrict__ out,
    int Bn)
{
    __shared__ float red[256];
    float s = 0.f;
    for (int i = threadIdx.x; i < Bn; i += 256) {
        float d2 = 0.f;
        #pragma unroll
        for (int sl = 0; sl < J_SLICES; ++sl)
            d2 += D2part[(size_t)sl * Bn + i];
        s += log1pf(d2 / D1[i]);
    }
    red[threadIdx.x] = s;
    __syncthreads();
    #pragma unroll
    for (int off = 128; off > 0; off >>= 1) {
        if (threadIdx.x < off) red[threadIdx.x] += red[threadIdx.x + off];
        __syncthreads();
    }
    if (threadIdx.x == 0) out[0] = red[0] / (float)Bn;
}

// ---------------------------------------------------------------------------
extern "C" void kernel_launch(void* const* d_in, const int* in_sizes, int n_in,
                              void* d_out, int out_size, void* d_ws, size_t ws_size,
                              hipStream_t stream)
{
    const float* anchor = (const float*)d_in[0];
    const float* pos    = (const float*)d_in[1];
    const float* neg    = (const float*)d_in[2];
    const int Bn = in_sizes[0] / D_DIM;            // 8192

    char* ws = (char*)d_ws;
    _Float16* a_hat  = (_Float16*)ws;                                  // Bn*256 f16
    _Float16* n_hat  = (_Float16*)(ws + (size_t)Bn * D_DIM * 2);       // Bn*256 f16
    float*    D1     = (float*)   (ws + (size_t)Bn * D_DIM * 4);       // Bn f32
    float*    D2part = D1 + Bn;                                        // J_SLICES*Bn f32

    tl_normalize_kernel<<<(Bn + 7) / 8, 256, 0, stream>>>(
        anchor, pos, neg, a_hat, n_hat, D1, Bn);

    dim3 grid(Bn / 128, J_SLICES);
    tl_gemm_exp_kernel<<<grid, 256, 0, stream>>>(
        a_hat, n_hat, D2part, Bn);

    tl_finalize_kernel<<<1, 256, 0, stream>>>(D1, D2part, (float*)d_out, Bn);
}
